// CausalMHA_71201967833252
// MI455X (gfx1250) — matrix operand order repaired and checked by reading, not compiled, not run on hardware
//
#include <hip/hip_runtime.h>
#include <hip/hip_bf16.h>
#include <cstdint>

#define B_  4
#define S_  2048
#define D_  1024
#define H_  16
#define DH_ 64

typedef __attribute__((ext_vector_type(16))) _Float16 v16h;
typedef __attribute__((ext_vector_type(8)))  _Float16 v8h;
typedef __attribute__((ext_vector_type(4)))  _Float16 v4h;
typedef __attribute__((ext_vector_type(8)))  float    v8f;
typedef int v4i __attribute__((vector_size(16)));

union F16Frag { v16h v; v8h h[2]; };

__device__ __forceinline__ v8f wmma_f16(const F16Frag& a, const F16Frag& b, v8f c) {
  // D = A(16x32 f16) * B(32x16 f16) + C(16x16 f32)
  return __builtin_amdgcn_wmma_f32_16x16x32_f16(false, a.v, false, b.v,
                                                (short)0, c, false, false);
}

#if __has_builtin(__builtin_amdgcn_global_load_async_to_lds_b128) && \
    __has_builtin(__builtin_amdgcn_s_wait_asynccnt)
#define USE_ASYNC_LDS 1
#else
#define USE_ASYNC_LDS 0
#endif

// copy 16 halves (32B) global -> LDS, async if available
__device__ __forceinline__ void g2l32B(const _Float16* g, _Float16* l) {
#if USE_ASYNC_LDS
  typedef __attribute__((address_space(1))) v4i as1_v4i;
  typedef __attribute__((address_space(3))) v4i as3_v4i;
  __builtin_amdgcn_global_load_async_to_lds_b128((as1_v4i*)g,       (as3_v4i*)l,       0, 0);
  __builtin_amdgcn_global_load_async_to_lds_b128((as1_v4i*)(g + 8), (as3_v4i*)(l + 8), 0, 0);
#else
  *(v8h*)l       = *(const v8h*)g;
  *(v8h*)(l + 8) = *(const v8h*)(g + 8);
#endif
}

__device__ __forceinline__ void async_tile_sync() {
#if USE_ASYNC_LDS
  __builtin_amdgcn_s_wait_asynccnt(0);
#endif
  __syncthreads();
}

// ---------------------------------------------------------------- fp32 -> fp16
__global__ __launch_bounds__(256) void k_f2h(const float* __restrict__ src,
                                             _Float16* __restrict__ dst, int n) {
  int i = (blockIdx.x * blockDim.x + threadIdx.x) * 4;
  if (i < n) {
    float4 f = *(const float4*)(src + i);
    v4h o = { (_Float16)f.x, (_Float16)f.y, (_Float16)f.z, (_Float16)f.w };
    *(v4h*)(dst + i) = o;
  }
}

// --------------------------------------------- QKV projection (+RoPE)
// 2x2 wmma register tile: one wave computes 32x32; block = 8 waves = 128x64 tile.
// mode 0: Q with RoPE  -> dst[((b*H+h)*S+s)*DH + d]
// mode 1: K with RoPE  -> same layout
// mode 2: V (no RoPE)  -> dst[((b*H+h)*DH+d)*S + s]   (transposed for PV B-fragments)
__global__ __launch_bounds__(256) void k_proj_qkv(const _Float16* __restrict__ xh,
                                                  const _Float16* __restrict__ wh,
                                                  _Float16* __restrict__ dst, int mode) {
  const int lane = threadIdx.x & 31;
  const int wid  = threadIdx.x >> 5;
  const int hi   = lane >> 4;
  const int l16  = lane & 15;
  const int wm   = wid >> 1, wn = wid & 1;
  const int m0   = blockIdx.x * 128 + wm * 32;
  const int n0   = blockIdx.y * 64  + wn * 32;

  const v8f zero = {0.f,0.f,0.f,0.f,0.f,0.f,0.f,0.f};
  v8f acc[2][2] = {{zero, zero}, {zero, zero}};

  const _Float16* ar0 = xh + (size_t)(m0 + l16) * D_;
  const _Float16* ar1 = ar0 + (size_t)16 * D_;
  const _Float16* br0 = wh + (size_t)(n0 + l16) * D_;  // W row-major [N,K] == B-frag order
  const _Float16* br1 = br0 + (size_t)16 * D_;

  for (int k0 = 0; k0 < D_; k0 += 32) {
    F16Frag a0, a1, b0, b1;
    a0.h[0] = *(const v8h*)(ar0 + k0 +      hi * 8);
    a0.h[1] = *(const v8h*)(ar0 + k0 + 16 + hi * 8);
    a1.h[0] = *(const v8h*)(ar1 + k0 +      hi * 8);
    a1.h[1] = *(const v8h*)(ar1 + k0 + 16 + hi * 8);
    b0.h[0] = *(const v8h*)(br0 + k0 +      hi * 8);
    b0.h[1] = *(const v8h*)(br0 + k0 + 16 + hi * 8);
    b1.h[0] = *(const v8h*)(br1 + k0 +      hi * 8);
    b1.h[1] = *(const v8h*)(br1 + k0 + 16 + hi * 8);
    acc[0][0] = wmma_f16(a0, b0, acc[0][0]);
    acc[0][1] = wmma_f16(a0, b1, acc[0][1]);
    acc[1][0] = wmma_f16(a1, b0, acc[1][0]);
    acc[1][1] = wmma_f16(a1, b1, acc[1][1]);
  }

  #pragma unroll
  for (int j = 0; j < 2; ++j) {
    const int n = n0 + j * 16 + l16;
    const int h = n >> 6, d = n & 63;
    const float inv_freq = __powf(10000.0f, -(float)(d & ~1) / 64.0f);
    const bool even = (d & 1) == 0;
    #pragma unroll
    for (int i = 0; i < 2; ++i) {
      #pragma unroll
      for (int r = 0; r < 8; ++r) {
        const int m  = m0 + i * 16 + r + hi * 8;
        const int bb = m >> 11;           // / S_
        const int s  = m & (S_ - 1);
        float val = acc[i][j][r];
        if (mode < 2) {                   // RoPE: pair (2i,2i+1) in adjacent lanes
          const float part = __shfl_xor(val, 1, 32);
          const float ang  = (float)s * inv_freq;
          const float cs = __cosf(ang), sn = __sinf(ang);
          val = even ? (val * cs - part * sn) : (part * sn + val * cs);
        }
        if (mode == 2)
          dst[((size_t)(bb * H_ + h) * DH_ + d) * S_ + s] = (_Float16)val;
        else
          dst[((size_t)(bb * H_ + h) * S_ + s) * DH_ + d] = (_Float16)val;
      }
    }
  }
}

// --------------------------------------------- flash attention
// Block = 4 waves, 64 q-rows, one (b,h). K/V tiles (32 keys) streamed through
// LDS double-buffered with async global->LDS copies, shared by all 4 waves.
__global__ __launch_bounds__(128) void k_attn(const _Float16* __restrict__ Qh,
                                              const _Float16* __restrict__ Kh,
                                              const _Float16* __restrict__ Vt,
                                              _Float16* __restrict__ Ah) {
  __shared__ __attribute__((aligned(32))) _Float16 Kl[2][32][64];  // [buf][key][dh]
  __shared__ __attribute__((aligned(32))) _Float16 Vl[2][64][32];  // [buf][dh][key]
  __shared__ __attribute__((aligned(32))) _Float16 Pt[4][16][32];  // per-wave P tile

  const int tid  = threadIdx.x;
  const int lane = tid & 31;
  const int wid  = tid >> 5;
  const int hi   = lane >> 4;
  const int l16  = lane & 15;
  const int bh   = blockIdx.y;
  const int b    = bh >> 4, h = bh & 15;
  const int q0   = blockIdx.x * 64 + wid * 16;
  const int nsteps = (blockIdx.x + 1) * 2;          // uniform causal bound for block

  const _Float16* Qb = Qh + (size_t)bh * S_ * DH_;
  const _Float16* Kb = Kh + (size_t)bh * S_ * DH_;
  const _Float16* Vb = Vt + (size_t)bh * DH_ * S_;

  // Q A-fragments (rows q0+l16, dh split 0..31 / 32..63)
  F16Frag qa[2];
  const _Float16* qrow = Qb + (size_t)(q0 + l16) * DH_;
  #pragma unroll
  for (int t = 0; t < 2; ++t) {
    qa[t].h[0] = *(const v8h*)(qrow + t * 32 +      hi * 8);
    qa[t].h[1] = *(const v8h*)(qrow + t * 32 + 16 + hi * 8);
  }

  const v8f zero = {0.f,0.f,0.f,0.f,0.f,0.f,0.f,0.f};
  v8f O[4] = {zero, zero, zero, zero};
  float mrow[8], lrow[8];
  #pragma unroll
  for (int r = 0; r < 8; ++r) { mrow[r] = -1e30f; lrow[r] = 0.f; }

  const float scale = 0.125f;          // 1/sqrt(64)

  // cooperative K/V tile fill (async -> LDS); 128 threads, 32B per thread per tile
  const int krow = tid >> 2, koff = (tid & 3) << 4;   // K: 32 rows x 128B
  const int vrow = tid >> 1, voff = (tid & 1) << 4;   // V: 64 rows x  64B
  auto load_tiles = [&](int st, int bf) {
    const int kt = st * 32;
    g2l32B(Kb + (size_t)(kt + krow) * DH_ + koff, &Kl[bf][krow][koff]);
    g2l32B(Vb + (size_t)vrow * S_ + kt + voff,    &Vl[bf][vrow][voff]);
  };

  load_tiles(0, 0);
  async_tile_sync();

  for (int st = 0; st < nsteps; ++st) {
    const int bf = st & 1;
    if (st + 1 < nsteps) load_tiles(st + 1, bf ^ 1);   // prefetch next tile
    const int kt = st * 32;

    // ---- scores from LDS K tile
    v8f s0 = zero, s1 = zero;
    #pragma unroll
    for (int t = 0; t < 2; ++t) {
      F16Frag kb;
      kb.h[0] = *(const v8h*)(&Kl[bf][l16][t * 32 + hi * 8]);
      kb.h[1] = *(const v8h*)(&Kl[bf][l16][t * 32 + 16 + hi * 8]);
      s0 = wmma_f16(qa[t], kb, s0);
    }
    #pragma unroll
    for (int t = 0; t < 2; ++t) {
      F16Frag kb;
      kb.h[0] = *(const v8h*)(&Kl[bf][16 + l16][t * 32 + hi * 8]);
      kb.h[1] = *(const v8h*)(&Kl[bf][16 + l16][t * 32 + 16 + hi * 8]);
      s1 = wmma_f16(qa[t], kb, s1);
    }

    // ---- online softmax
    const int key0 = kt + l16, key1 = key0 + 16;
    #pragma unroll
    for (int r = 0; r < 8; ++r) {
      const int q = q0 + r + hi * 8;
      float v0 = (key0 <= q) ? s0[r] * scale : -1e30f;
      float v1 = (key1 <= q) ? s1[r] * scale : -1e30f;
      float mx = fmaxf(v0, v1);
      #pragma unroll
      for (int msk = 1; msk < 16; msk <<= 1) mx = fmaxf(mx, __shfl_xor(mx, msk, 32));
      const float mnew  = fmaxf(mrow[r], mx);
      const float alpha = __expf(mrow[r] - mnew);
      const float p0 = __expf(v0 - mnew);
      const float p1 = __expf(v1 - mnew);
      float sum = p0 + p1;
      #pragma unroll
      for (int msk = 1; msk < 16; msk <<= 1) sum += __shfl_xor(sum, msk, 32);
      lrow[r] = lrow[r] * alpha + sum;
      mrow[r] = mnew;
      #pragma unroll
      for (int f = 0; f < 4; ++f) O[f][r] *= alpha;
      const int row = r + hi * 8;
      Pt[wid][row][l16]      = (_Float16)p0;
      Pt[wid][row][16 + l16] = (_Float16)p1;
    }

    // ---- P as A-fragment (per-wave LDS region; same-wave DS ops are in order)
    F16Frag pa;
    pa.h[0] = *(const v8h*)(&Pt[wid][l16][hi * 8]);
    pa.h[1] = *(const v8h*)(&Pt[wid][l16][16 + hi * 8]);

    // ---- PV from LDS V tile (transposed: contiguous along keys)
    #pragma unroll
    for (int f = 0; f < 4; ++f) {
      F16Frag vbf;
      vbf.h[0] = *(const v8h*)(&Vl[bf][f * 16 + l16][hi * 8]);
      vbf.h[1] = *(const v8h*)(&Vl[bf][f * 16 + l16][16 + hi * 8]);
      O[f] = wmma_f16(pa, vbf, O[f]);
    }

    async_tile_sync();   // next tile landed; everyone done reading buf bf
  }

  // ---- normalize + store fp16 [B*S, D] for the output projection
  #pragma unroll
  for (int r = 0; r < 8; ++r) {
    const float inv = 1.0f / lrow[r];
    const int s = q0 + r + hi * 8;
    _Float16* orow = Ah + (size_t)(b * S_ + s) * D_ + h * 64;
    #pragma unroll
    for (int f = 0; f < 4; ++f)
      orow[f * 16 + l16] = (_Float16)(O[f][r] * inv);
  }
}

// --------------------------------------------- output projection (2x2 tile), fp32 store
__global__ __launch_bounds__(256) void k_proj_out(const _Float16* __restrict__ ah,
                                                  const _Float16* __restrict__ wh,
                                                  float* __restrict__ out) {
  const int lane = threadIdx.x & 31;
  const int wid  = threadIdx.x >> 5;
  const int hi   = lane >> 4;
  const int l16  = lane & 15;
  const int wm   = wid >> 1, wn = wid & 1;
  const int m0   = blockIdx.x * 128 + wm * 32;
  const int n0   = blockIdx.y * 64  + wn * 32;

  const v8f zero = {0.f,0.f,0.f,0.f,0.f,0.f,0.f,0.f};
  v8f acc[2][2] = {{zero, zero}, {zero, zero}};

  const _Float16* ar0 = ah + (size_t)(m0 + l16) * D_;
  const _Float16* ar1 = ar0 + (size_t)16 * D_;
  const _Float16* br0 = wh + (size_t)(n0 + l16) * D_;
  const _Float16* br1 = br0 + (size_t)16 * D_;

  for (int k0 = 0; k0 < D_; k0 += 32) {
    F16Frag a0, a1, b0, b1;
    a0.h[0] = *(const v8h*)(ar0 + k0 +      hi * 8);
    a0.h[1] = *(const v8h*)(ar0 + k0 + 16 + hi * 8);
    a1.h[0] = *(const v8h*)(ar1 + k0 +      hi * 8);
    a1.h[1] = *(const v8h*)(ar1 + k0 + 16 + hi * 8);
    b0.h[0] = *(const v8h*)(br0 + k0 +      hi * 8);
    b0.h[1] = *(const v8h*)(br0 + k0 + 16 + hi * 8);
    b1.h[0] = *(const v8h*)(br1 + k0 +      hi * 8);
    b1.h[1] = *(const v8h*)(br1 + k0 + 16 + hi * 8);
    acc[0][0] = wmma_f16(a0, b0, acc[0][0]);
    acc[0][1] = wmma_f16(a0, b1, acc[0][1]);
    acc[1][0] = wmma_f16(a1, b0, acc[1][0]);
    acc[1][1] = wmma_f16(a1, b1, acc[1][1]);
  }

  #pragma unroll
  for (int i = 0; i < 2; ++i)
    #pragma unroll
    for (int j = 0; j < 2; ++j)
      #pragma unroll
      for (int r = 0; r < 8; ++r)
        out[(size_t)(m0 + i * 16 + r + hi * 8) * D_ + n0 + j * 16 + l16] = acc[i][j][r];
}

// ----------------------------------------------------------------------------
extern "C" void kernel_launch(void* const* d_in, const int* in_sizes, int n_in,
                              void* d_out, int out_size, void* d_ws, size_t ws_size,
                              hipStream_t stream) {
  const float* x  = (const float*)d_in[0];
  const float* wq = (const float*)d_in[1];
  const float* wk = (const float*)d_in[2];
  const float* wv = (const float*)d_in[3];
  const float* wo = (const float*)d_in[4];

  const size_t NX = (size_t)B_ * S_ * D_;   // 8,388,608
  const size_t NW = (size_t)D_ * D_;        // 1,048,576

  char* ws = (char*)d_ws;
  size_t off = 0;
  _Float16* xh  = (_Float16*)(ws + off); off += NX * 2;
  _Float16* wqh = (_Float16*)(ws + off); off += NW * 2;
  _Float16* wkh = (_Float16*)(ws + off); off += NW * 2;
  _Float16* wvh = (_Float16*)(ws + off); off += NW * 2;
  _Float16* woh = (_Float16*)(ws + off); off += NW * 2;
  _Float16* Qh  = (_Float16*)(ws + off); off += NX * 2;
  _Float16* Kh  = (_Float16*)(ws + off); off += NX * 2;
  _Float16* Vt  = (_Float16*)(ws + off); off += NX * 2;
  _Float16* Ah  = (_Float16*)(ws + off); off += NX * 2;
  if (ws_size < off) return;

  k_f2h<<<(int)((NX / 4 + 255) / 256), 256, 0, stream>>>(x,  xh,  (int)NX);
  k_f2h<<<(int)((NW / 4 + 255) / 256), 256, 0, stream>>>(wq, wqh, (int)NW);
  k_f2h<<<(int)((NW / 4 + 255) / 256), 256, 0, stream>>>(wk, wkh, (int)NW);
  k_f2h<<<(int)((NW / 4 + 255) / 256), 256, 0, stream>>>(wv, wvh, (int)NW);
  k_f2h<<<(int)((NW / 4 + 255) / 256), 256, 0, stream>>>(wo, woh, (int)NW);

  dim3 gproj(64, 16);      // M: 8192/128, N: 1024/64
  k_proj_qkv<<<gproj, 256, 0, stream>>>(xh, wqh, Qh, 0);
  k_proj_qkv<<<gproj, 256, 0, stream>>>(xh, wkh, Kh, 1);
  k_proj_qkv<<<gproj, 256, 0, stream>>>(xh, wvh, Vt, 2);

  dim3 gattn(S_ / 64, B_ * H_);   // (32, 64), 4 waves/block
  k_attn<<<gattn, 128, 0, stream>>>(Qh, Kh, Vt, Ah);

  k_proj_out<<<gproj, 256, 0, stream>>>(Ah, woh, (float*)d_out);
}
